// DMGCGRUCell_27298812133880
// MI455X (gfx1250) — matrix operand order repaired and checked by reading, not compiled, not run on hardware
//
#include <hip/hip_runtime.h>
#include <hip/hip_bf16.h>
#include <stdint.h>

// Problem constants (match reference)
#define B_    16
#define N_    10000
#define R_    10
#define S_    1000
#define G_    4
#define DIN_  32
#define DH_   64
#define DBLK_ 96
#define SP_   1024   // S padded to multiple of 32 for WMMA K-steps / 16 for M-tiles

typedef __attribute__((ext_vector_type(16))) __bf16 v16bf;
typedef __attribute__((ext_vector_type(8)))  float  v8f;
typedef __attribute__((ext_vector_type(4)))  unsigned int v4u;
typedef __attribute__((ext_vector_type(8)))  int    v8i_;
typedef __attribute__((ext_vector_type(4)))  int    v4i_;

struct __align__(16) U128 { unsigned int a, b, c, d; };
union Frag32 { v16bf v; U128 u[2]; };   // 32-byte WMMA 16-bit fragment

__device__ __forceinline__ uint16_t f2bf(float f) {
  uint32_t x = __float_as_uint(f);
  uint32_t r = x + 0x7FFFu + ((x >> 16) & 1u);   // round-to-nearest-even
  return (uint16_t)(r >> 16);
}

// ---------------------------------------------------------------- kernel 1
// deg[g*R+r, s] = rsqrt(max(sum_t A, 1e-5)); one wave32 per row.
__global__ __launch_bounds__(256) void k_deg(const float* __restrict__ A,
                                             float* __restrict__ deg) {
  int wave = (int)((blockIdx.x * blockDim.x + threadIdx.x) >> 5);
  int lane = threadIdx.x & 31;
  if (wave >= G_ * R_ * S_) return;
  const float* row = A + (size_t)wave * S_;
  float s = 0.f;
  for (int t = lane; t < S_; t += 32) s += row[t];
  for (int off = 16; off > 0; off >>= 1) s += __shfl_xor(s, off, 32);
  if (lane == 0) deg[wave] = rsqrtf(fmaxf(s, 1e-5f));
}

// ---------------------------------------------------------------- kernel 2
// An[gr][s][t] (bf16, padded SPxSP) = deg_s * (A + I) * deg_t ; zero padding.
__global__ __launch_bounds__(256) void k_an(const float* __restrict__ A,
                                            const float* __restrict__ deg,
                                            uint16_t* __restrict__ An) {
  size_t idx = (size_t)blockIdx.x * blockDim.x + threadIdx.x;
  const size_t total = (size_t)G_ * R_ * SP_ * SP_;
  if (idx >= total) return;
  int t  = (int)(idx & (SP_ - 1));
  int s  = (int)((idx >> 10) & (SP_ - 1));
  int gr = (int)(idx >> 20);                       // SP*SP == 2^20
  float v = 0.f;
  if (s < S_ && t < S_) {
    float a = A[((size_t)gr * S_ + s) * S_ + t] + (s == t ? 1.f : 0.f);
    v = deg[gr * S_ + s] * a * deg[gr * S_ + t];
  }
  An[idx] = f2bf(v);
}

// ---------------------------------------------------------------- kernel 3
// inp bf16 [B][R][SP][96]: mode 0 -> [x_t | h_prev], mode 1 -> [x_t | Rg*h_prev]
__global__ __launch_bounds__(256) void k_pack(const float* __restrict__ x,
                                              const float* __restrict__ h,
                                              const float* __restrict__ Rg,
                                              int mode,
                                              uint16_t* __restrict__ inp) {
  size_t idx = (size_t)blockIdx.x * blockDim.x + threadIdx.x;
  const size_t total = (size_t)B_ * R_ * SP_ * DBLK_;
  if (idx >= total) return;
  int d = (int)(idx % DBLK_);
  size_t row = idx / DBLK_;
  int t  = (int)(row & (SP_ - 1));
  int br = (int)(row >> 10);
  int r = br % R_, b = br / R_;
  float v = 0.f;
  if (t < S_) {
    size_t bn = (size_t)b * N_ + (size_t)r * S_ + t;
    if (d < DIN_) v = x[bn * DIN_ + d];
    else {
      float hv = h[bn * DH_ + (d - DIN_)];
      v = (mode == 0) ? hv : Rg[bn * DH_ + (d - DIN_)] * hv;
    }
  }
  inp[idx] = f2bf(v);
}

// ---------------------------------------------------------------- kernel 4
// hwT[g][b*R+r][e][t] = (inp @ W[g]) transposed, bf16. WMMA M-tile per wave.
#define MT4 ((B_ * R_ * SP_) / 16)   // 10240 M-tiles
__global__ __launch_bounds__(256) void k_hw(const uint16_t* __restrict__ inp,
                                            const float* __restrict__ W,
                                            uint16_t* __restrict__ hwT) {
  __shared__ __align__(16) uint16_t Wl[DH_ * DBLK_];  // col-major [e][d], 12 KB
  const int BLOCKS_PER_G = MT4 / 8;                   // 1280
  int g    = blockIdx.x / BLOCKS_PER_G;
  int mblk = blockIdx.x % BLOCKS_PER_G;
  for (int i = threadIdx.x; i < DH_ * DBLK_; i += 256) {
    int e = i / DBLK_, d = i % DBLK_;
    Wl[i] = f2bf(W[((size_t)g * DBLK_ + d) * DH_ + e]);
  }
  __syncthreads();
  int wave = threadIdx.x >> 5, lane = threadIdx.x & 31;
  int mt = mblk * 8 + wave;
  int half = lane >> 4, l15 = lane & 15;
  int koffA = half * 16;   // A frag: per-lane rows, 16 contiguous K
  int koffB = half * 16;   // B frag: per-lane cols, 16 contiguous K
  int rowA = mt * 16 + l15;
  const uint16_t* arow = inp + (size_t)rowA * DBLK_;
  v8f acc[4] = {};
  for (int ks = 0; ks < 3; ++ks) {
    int k0 = ks * 32;
    Frag32 a;
    a.u[0] = *(const U128*)(arow + k0 + koffA);
    a.u[1] = *(const U128*)(arow + k0 + koffA + 8);
    for (int nt = 0; nt < 4; ++nt) {
      Frag32 bm;
      const uint16_t* bp = Wl + (nt * 16 + l15) * DBLK_ + k0 + koffB;
      bm.u[0] = *(const U128*)(bp);
      bm.u[1] = *(const U128*)(bp + 8);
      acc[nt] = __builtin_amdgcn_wmma_f32_16x16x32_bf16(
          false, a.v, false, bm.v, (short)0, acc[nt], false, false);
    }
  }
  for (int nt = 0; nt < 4; ++nt)
    for (int i = 0; i < 8; ++i) {
      int m = mt * 16 + (half ? 8 + i : i);          // global input row
      int t = m & (SP_ - 1);
      int br = m >> 10;                               // b*R + r
      int e = nt * 16 + l15;
      size_t off = (((size_t)g * (B_ * R_) + br) * DH_ + e) * SP_ + t;
      hwT[off] = f2bf(acc[nt][i]);
    }
}

// ---------------------------------------------------------------- kernel 5
// outs[g][b][n][e] = relu(An[g,r] @ hw[g,b,r]); per (g,b,r): 1024x64, K=1024.
// 128 KB hwT slab pulled into LDS by the Tensor Data Mover (one D#, one
// tensor_load_to_lds, TENSORcnt-tracked). 8 waves x 2 M-tiles, double-buffered
// A (global) and B (LDS) fragments so WMMA overlaps the next K-step's loads.
__global__ __launch_bounds__(256) void k_gemm(const uint16_t* __restrict__ An,
                                              const uint16_t* __restrict__ hwT,
                                              float* __restrict__ outs) {
  extern __shared__ uint16_t lds[];                  // [64][1024] bf16 = 131072 B
  int mb  = blockIdx.x & 3;
  int gbr = blockIdx.x >> 2;                          // (g*B + b)*R + r
  int r = gbr % R_;
  int gb = gbr / R_;
  int b = gb % B_, g = gb / B_;
  const uint16_t* src = hwT + (size_t)gbr * (DH_ * SP_);

  // --- TDM slab load: one descriptor, 16384 x 8B contiguous tile -> LDS.
  {
    uint64_t ga = (uint64_t)(uintptr_t)src;
    uint32_t lds_base = __builtin_amdgcn_groupstaticsize(); // dynamic-LDS offset
    v4u g0;
    g0.x = 0x1u;                                     // count=1, user descriptor
    g0.y = lds_base;                                 // lds_addr[31:0]
    g0.z = (uint32_t)ga;                             // global_addr[31:0]
    g0.w = (uint32_t)((ga >> 32) & 0x01FFFFFFu) | (0x2u << 30); // addr[56:32],type=2
    v8i_ g1;
    g1[0] = 0x00030000;          // workgroup_mask=0, data_size=3 (8 bytes)
    g1[1] = (int)0x40000000;     // tensor_dim0[15:0]=0x4000 in bits 63:48
    g1[2] = 0x00010000;          // tensor_dim0 hi=0 | tensor_dim1[15:0]=1
    g1[3] = (int)0x40000000;     // tensor_dim1 hi=0 | tile_dim0=0x4000
    g1[4] = 0x00000001;          // tile_dim1=1, tile_dim2=0
    g1[5] = 0x00004000;          // tensor_dim0_stride = 16384 (low 32)
    g1[6] = 0;
    g1[7] = 0;
    v4i_ gz = {0, 0, 0, 0};
    v8i_ gz8 = {0, 0, 0, 0, 0, 0, 0, 0};
    if (threadIdx.x < 32) {                          // one wave issues the DMA
      __builtin_amdgcn_tensor_load_to_lds(g0, g1, gz, gz, gz8, 0);
      __builtin_amdgcn_s_wait_tensorcnt(0);          // s_wait_tensorcnt 0x0
    }
    __syncthreads();                                 // release the other waves
  }

  int wave = threadIdx.x >> 5, lane = threadIdx.x & 31;
  int half = lane >> 4, l15 = lane & 15;
  int koffA = half * 16, koffB = half * 16;
  int mt0 = mb * 16 + wave * 2;                       // two M-tiles per wave
  const uint16_t* anb   = An + ((size_t)g * R_ + r) * SP_ * SP_;
  const uint16_t* arow0 = anb + (size_t)(mt0 * 16 + l15) * SP_;
  const uint16_t* arow1 = arow0 + (size_t)16 * SP_;
  v8f acc[2][4] = {};
  Frag32 bfr[2][4], afr[2][2];

#define LOAD_B(kk, buf)                                                     \
  do {                                                                      \
    int k0_ = (kk) * 32;                                                    \
    for (int nt = 0; nt < 4; ++nt) {                                        \
      const uint16_t* bp = lds + ((nt * 16 + l15) * SP_ + k0_ + koffB);     \
      bfr[buf][nt].u[0] = *(const U128*)bp;                                 \
      bfr[buf][nt].u[1] = *(const U128*)(bp + 8);                           \
    }                                                                       \
  } while (0)
#define LOAD_A(kk, buf)                                                     \
  do {                                                                      \
    int k0_ = (kk) * 32;                                                    \
    afr[buf][0].u[0] = *(const U128*)(arow0 + k0_ + koffA);                 \
    afr[buf][0].u[1] = *(const U128*)(arow0 + k0_ + koffA + 8);             \
    afr[buf][1].u[0] = *(const U128*)(arow1 + k0_ + koffA);                 \
    afr[buf][1].u[1] = *(const U128*)(arow1 + k0_ + koffA + 8);             \
  } while (0)

  LOAD_B(0, 0);
  LOAD_A(0, 0);
  for (int kk = 0; kk < 32; ++kk) {
    int cur = kk & 1, nxt = cur ^ 1;
    if (kk + 1 < 32) {                                // prefetch next K-step
      LOAD_B(kk + 1, nxt);
      LOAD_A(kk + 1, nxt);
    }
    for (int nt = 0; nt < 4; ++nt) {
      acc[0][nt] = __builtin_amdgcn_wmma_f32_16x16x32_bf16(
          false, afr[cur][0].v, false, bfr[cur][nt].v, (short)0, acc[0][nt],
          false, false);
      acc[1][nt] = __builtin_amdgcn_wmma_f32_16x16x32_bf16(
          false, afr[cur][1].v, false, bfr[cur][nt].v, (short)0, acc[1][nt],
          false, false);
    }
  }
#undef LOAD_B
#undef LOAD_A

  float* ob = outs + (((size_t)g * B_ + b) * N_ + (size_t)r * S_) * DH_;
  for (int m = 0; m < 2; ++m)
    for (int nt = 0; nt < 4; ++nt)
      for (int i = 0; i < 8; ++i) {
        int s = (mt0 + m) * 16 + (half ? 8 + i : i);
        if (s < S_) {
          int e = nt * 16 + l15;
          ob[(size_t)s * DH_ + e] = fmaxf(acc[m][nt][i], 0.f);
        }
      }
}

// ---------------------------------------------------------------- kernel 6
// Attention MLP + softmax + residual bias + blend; gate c fuses GRU combine.
__global__ __launch_bounds__(256) void k_attn(const float* __restrict__ outs,
                                              const float* __restrict__ rs,
                                              const float* __restrict__ a1w,
                                              const float* __restrict__ a1b,
                                              const float* __restrict__ a2w,
                                              const float* __restrict__ a2b,
                                              const float* __restrict__ U,
                                              const float* __restrict__ hprev,
                                              float* __restrict__ dst, int mode) {
  extern __shared__ float sm[];                       // a1w + a1b + a2w
  float* s_a1w = sm;                                  // 258*64
  float* s_a1b = sm + 258 * 64;
  float* s_a2w = s_a1b + 64;                          // 64*4
  for (int i = threadIdx.x; i < 258 * 64; i += 256) s_a1w[i] = a1w[i];
  if (threadIdx.x < 64) s_a1b[threadIdx.x] = a1b[threadIdx.x];
  for (int i = threadIdx.x; i < 64 * G_; i += 256) s_a2w[i] = a2w[i];
  __syncthreads();
  size_t idx = (size_t)blockIdx.x * 256 + threadIdx.x;
  if (idx >= (size_t)B_ * N_) return;
  int b = (int)(idx / N_), n = (int)(idx % N_);
  float rs0 = rs[idx * 2 + 0], rs1 = rs[idx * 2 + 1];
  float s2[4] = {a2b[0], a2b[1], a2b[2], a2b[3]};
  for (int jc = 0; jc < 4; ++jc) {                    // 16-wide j blocks
    float acc[16];
    for (int jj = 0; jj < 16; ++jj) acc[jj] = s_a1b[jc * 16 + jj];
    for (int g = 0; g < G_; ++g) {
      const float* hp = outs + (((size_t)g * B_ + b) * N_ + n) * DH_;
      for (int e = 0; e < DH_; ++e) {
        float hd = hp[e];
        const float* wr = s_a1w + (g * DH_ + e) * 64 + jc * 16;
        for (int jj = 0; jj < 16; ++jj) acc[jj] += hd * wr[jj];
      }
    }
    const float* wr = s_a1w + 256 * 64 + jc * 16;
    for (int jj = 0; jj < 16; ++jj) acc[jj] += rs0 * wr[jj] + rs1 * wr[64 + jj];
    for (int jj = 0; jj < 16; ++jj) {
      float z = fmaxf(acc[jj], 0.f);
      int j = jc * 16 + jj;
      for (int g = 0; g < G_; ++g) s2[g] += z * s_a2w[j * 4 + g];
    }
  }
  float mx = fmaxf(fmaxf(s2[0], s2[1]), fmaxf(s2[2], s2[3]));
  float w[4], sw = 0.f;
  for (int g = 0; g < 4; ++g) { w[g] = __expf(s2[g] - mx); sw += w[g]; }
  const float BIASW[4] = {0.1f, 0.1f, 0.1f, 1.0f};
  float flag = (rs1 > 0.5f) ? 1.f : 0.f, sa = 0.f;
  for (int g = 0; g < 4; ++g) { w[g] = (w[g] / sw) * (1.f + flag * BIASW[g]); sa += w[g]; }
  float inv = 1.f / sa;
  for (int g = 0; g < 4; ++g) w[g] *= inv;
  float* out = dst + idx * DH_;
  for (int e = 0; e < DH_; ++e) {
    float v = 0.f;
    for (int g = 0; g < 4; ++g)
      v += w[g] * outs[(((size_t)g * B_ + b) * N_ + n) * DH_ + e];
    if (mode == 2) {
      float u = 1.f / (1.f + __expf(-U[idx * DH_ + e]));
      v = u * tanhf(v) + (1.f - u) * hprev[idx * DH_ + e];
    }
    out[e] = v;
  }
}

// ---------------------------------------------------------------- driver
extern "C" void kernel_launch(void* const* d_in, const int* in_sizes, int n_in,
                              void* d_out, int out_size, void* d_ws, size_t ws_size,
                              hipStream_t stream) {
  (void)in_sizes; (void)n_in; (void)out_size; (void)ws_size;
  const float* x_t  = (const float*)d_in[0];
  const float* h_pr = (const float*)d_in[1];
  const float* A    = (const float*)d_in[2];
  const float* rs   = (const float*)d_in[3];
  const float* Wt[3]   = {(const float*)d_in[4],  (const float*)d_in[9],  (const float*)d_in[14]};
  const float* a1wt[3] = {(const float*)d_in[5],  (const float*)d_in[10], (const float*)d_in[15]};
  const float* a1bt[3] = {(const float*)d_in[6],  (const float*)d_in[11], (const float*)d_in[16]};
  const float* a2wt[3] = {(const float*)d_in[7],  (const float*)d_in[12], (const float*)d_in[17]};
  const float* a2bt[3] = {(const float*)d_in[8],  (const float*)d_in[13], (const float*)d_in[18]};

  // Workspace layout (all 256-B aligned); total ~426 MiB.
  char* ws = (char*)d_ws;
  size_t off = 0;
  float*    deg  = (float*)(ws + off);    off += (size_t)G_ * R_ * S_ * 4;          off = (off + 255) & ~(size_t)255;
  uint16_t* An   = (uint16_t*)(ws + off); off += (size_t)G_ * R_ * SP_ * SP_ * 2;   off = (off + 255) & ~(size_t)255;
  uint16_t* inp  = (uint16_t*)(ws + off); off += (size_t)B_ * R_ * SP_ * DBLK_ * 2; off = (off + 255) & ~(size_t)255;
  uint16_t* hwT  = (uint16_t*)(ws + off); off += (size_t)G_ * B_ * R_ * DH_ * SP_ * 2; off = (off + 255) & ~(size_t)255;
  float*    outs = (float*)(ws + off);    off += (size_t)G_ * B_ * N_ * DH_ * 4;    off = (off + 255) & ~(size_t)255;
  float*    Ubuf = (float*)(ws + off);    off += (size_t)B_ * N_ * DH_ * 4;         off = (off + 255) & ~(size_t)255;
  float*    Rbuf = (float*)(ws + off);

  const size_t attn_lds = (size_t)(258 * 64 + 64 + 256) * sizeof(float); // 67328
  const size_t gemm_lds = (size_t)DH_ * SP_ * 2;                          // 131072

  // Shared, gate-invariant normalized adjacency (bf16, L2-resident)
  k_deg<<<dim3((G_ * R_ * S_ * 32) / 256), dim3(256), 0, stream>>>(A, deg);
  k_an<<<dim3(((size_t)G_ * R_ * SP_ * SP_) / 256), dim3(256), 0, stream>>>(A, deg, An);

  // Gates u and r share the packed input [x_t | h_prev]
  k_pack<<<dim3(((size_t)B_ * R_ * SP_ * DBLK_) / 256), dim3(256), 0, stream>>>(x_t, h_pr, Rbuf, 0, inp);

  float* gate_dst[3] = {Ubuf, Rbuf, (float*)d_out};
  for (int gate = 0; gate < 3; ++gate) {
    if (gate == 2)  // candidate path: [x_t | Rg * h_prev]
      k_pack<<<dim3(((size_t)B_ * R_ * SP_ * DBLK_) / 256), dim3(256), 0, stream>>>(x_t, h_pr, Rbuf, 1, inp);
    k_hw<<<dim3(G_ * (MT4 / 8)), dim3(256), 0, stream>>>(inp, Wt[gate], hwT);
    k_gemm<<<dim3(G_ * B_ * R_ * 4), dim3(256), gemm_lds, stream>>>(An, hwT, outs);
    k_attn<<<dim3((B_ * N_) / 256), dim3(256), attn_lds, stream>>>(
        outs, rs, a1wt[gate], a1bt[gate], a2wt[gate], a2bt[gate],
        Ubuf, h_pr, gate_dst[gate], gate);
  }
}
